// DBGSR_1675037245687
// MI455X (gfx1250) — compile-verified, matrix operand order checked by reading, not run on hardware
//
#include <hip/hip_runtime.h>
#include <hip/hip_bf16.h>

// ---------------------------------------------------------------------------
// Types for CDNA5 WMMA (gfx1250, wave32)
// ---------------------------------------------------------------------------
typedef __attribute__((ext_vector_type(16))) __bf16 v16bf;
typedef __attribute__((ext_vector_type(8)))  float  v8f;

#define TAN30F 0.57735026918962576451f

static __host__ __device__ inline long lmax(long a, long b){ return a > b ? a : b; }
static inline int blks(long n){ return (int)((n + 255) / 256); }

// ---------------------------------------------------------------------------
// Utility kernels
// ---------------------------------------------------------------------------
__global__ void zero_f32_k(float* __restrict__ p, long n){
    long t = (long)blockIdx.x * blockDim.x + threadIdx.x;
    if (t < n) p[t] = 0.f;
}
__global__ void zero_bf16_k(__bf16* __restrict__ p, long n){
    long t = (long)blockIdx.x * blockDim.x + threadIdx.x;
    if (t < n) p[t] = (__bf16)0.f;
}

// dst[r*dld + doff + c] = (bf16) src[r*sld + soff + c]
__global__ void cvt_cols_k(const float* __restrict__ src, int sld, int soff,
                           __bf16* __restrict__ dst, int dld, int doff,
                           long rows, int cols){
    long t = (long)blockIdx.x * blockDim.x + threadIdx.x;
    if (t >= rows * (long)cols) return;
    long r = t / cols; int c = (int)(t - r * cols);
    dst[r * (long)dld + doff + c] = (__bf16)src[r * (long)sld + soff + c];
}

// Edge-parallel scatter-add: agg[dst[e]][coff+c] += x[src[e]][c]
__global__ void scatter_add_k(const float* __restrict__ x, int C,
                              const int* __restrict__ src, const int* __restrict__ dst,
                              long E, float* __restrict__ agg, int lda, int coff){
    long t = (long)blockIdx.x * blockDim.x + threadIdx.x;
    if (t >= E * (long)C) return;
    long e = t / C; int c = (int)(t - e * C);
    atomicAdd(&agg[(long)dst[e] * lda + coff + c], x[(long)src[e] * C + c]);
}

// Weight prep: WT[co*Kpad + k] = W[k*cout + co] (bf16, zero padded); bp = padded bias
__global__ void prep_weight_k(const float* __restrict__ W, const float* __restrict__ bias,
                              __bf16* __restrict__ WT, float* __restrict__ bp,
                              int cin, int cout, int Kpad, int COpad){
    int idx = blockIdx.x * blockDim.x + threadIdx.x;
    int tot = COpad * Kpad;
    if (idx < tot){
        int co = idx / Kpad, k = idx - co * Kpad;
        float v = (co < cout && k < cin) ? W[(long)k * cout + co] : 0.f;
        WT[idx] = (__bf16)v;
    }
    if (idx < COpad) bp[idx] = (idx < cout) ? bias[idx] : 0.f;
}

// h = concat(x[4], pos2[3], y*ones[3]) -> f32 [N2,10] and bf16 [N2,32] (zero padded)
__global__ void build_h_k(const float* __restrict__ x, const float* __restrict__ pos,
                          const float* __restrict__ y, float* __restrict__ h,
                          __bf16* __restrict__ hb, int n){
    long t = (long)blockIdx.x * blockDim.x + threadIdx.x;
    if (t >= (long)n * 32) return;
    int r = (int)(t >> 5), c = (int)(t & 31);
    float v = 0.f;
    if (c < 4)       v = x[(long)r * 4 + c];
    else if (c < 7)  v = pos[(long)r * 3 + (c - 4)];
    else if (c < 10) v = y[0];
    if (c < 10) h[(long)r * 10 + c] = v;
    hb[t] = (__bf16)v;
}

// BatchNorm stats (training mode, biased variance): one block per channel
__global__ void bn_stats_k(const float* __restrict__ x, long N, int C,
                           const float* __restrict__ g, const float* __restrict__ b,
                           float* __restrict__ ss){
    int c = blockIdx.x;
    float s = 0.f, s2 = 0.f;
    for (long i = threadIdx.x; i < N; i += blockDim.x){
        float v = x[i * C + c]; s += v; s2 += v * v;
    }
    __shared__ float sh1[256], sh2[256];
    sh1[threadIdx.x] = s; sh2[threadIdx.x] = s2; __syncthreads();
    for (int w = 128; w > 0; w >>= 1){
        if ((int)threadIdx.x < w){
            sh1[threadIdx.x] += sh1[threadIdx.x + w];
            sh2[threadIdx.x] += sh2[threadIdx.x + w];
        }
        __syncthreads();
    }
    if (threadIdx.x == 0){
        float m   = sh1[0] / (float)N;
        float var = sh2[0] / (float)N - m * m;
        float sc  = g[c] * rsqrtf(var + 1e-5f);
        ss[c * 2]     = sc;
        ss[c * 2 + 1] = b[c] - m * sc;
    }
}

// y = (elu?)(x*scale+shift); optional f32 out, optional bf16 write at column offset
__global__ void bn_apply_k(const float* __restrict__ x, const float* __restrict__ ss,
                           float* __restrict__ of, __bf16* __restrict__ ob,
                           int obld, int oboff, long N, int C, int elu){
    long t = (long)blockIdx.x * blockDim.x + threadIdx.x;
    if (t >= N * (long)C) return;
    long r = t / C; int c = (int)(t - r * C);
    float v = x[t] * ss[c * 2] + ss[c * 2 + 1];
    if (elu && v < 0.f) v = expm1f(v);
    if (of) of[t] = v;
    if (ob) ob[r * (long)obld + oboff + c] = (__bf16)v;
}

__global__ void axpy_k(const float* __restrict__ a, const float* __restrict__ b,
                       float al, float* __restrict__ o, long n){
    long t = (long)blockIdx.x * blockDim.x + threadIdx.x;
    if (t < n) o[t] = a[t] + al * b[t];
}

// onera_transform: p0 = x - tan30*y; scale = 1 + (1/0.56-1)*(y/1.1963); out = (p0,y,z)*scale
__global__ void onera_k(const float* __restrict__ p, float* __restrict__ o, int n){
    int i = blockIdx.x * blockDim.x + threadIdx.x;
    if (i >= n) return;
    float x = p[(long)i * 3], y = p[(long)i * 3 + 1], z = p[(long)i * 3 + 2];
    float p0 = x - TAN30F * y;
    float s  = 1.f + (1.f / 0.56f - 1.f) * (y / 1.1963f);
    o[(long)i * 3]     = p0 * s;
    o[(long)i * 3 + 1] = y * s;
    o[(long)i * 3 + 2] = z * s;
}

// Brute-force 1-NN with LDS tiling of the candidate set
__global__ void knn1_k(const float* __restrict__ tp2, int n2,
                       const float* __restrict__ tp3, int n3,
                       int* __restrict__ idx){
    int q = blockIdx.x * blockDim.x + threadIdx.x;
    float qx = 0.f, qy = 0.f, qz = 0.f;
    if (q < n3){ qx = tp3[(long)q*3]; qy = tp3[(long)q*3+1]; qz = tp3[(long)q*3+2]; }
    __shared__ float sx[256], sy[256], sz[256];
    float best = 3.4e38f; int bi = 0;
    for (int base = 0; base < n2; base += 256){
        int j = base + threadIdx.x;
        if (j < n2){
            sx[threadIdx.x] = tp2[(long)j*3];
            sy[threadIdx.x] = tp2[(long)j*3+1];
            sz[threadIdx.x] = tp2[(long)j*3+2];
        }
        __syncthreads();
        int lim = (n2 - base) < 256 ? (n2 - base) : 256;
        #pragma unroll 4
        for (int t = 0; t < lim; ++t){
            float dx = qx - sx[t], dy = qy - sy[t], dz = qz - sz[t];
            float d = dx*dx + dy*dy + dz*dz;
            if (d < best){ best = d; bi = base + t; }  // first-min tie break (argmin)
        }
        __syncthreads();
    }
    if (q < n3) idx[q] = bi;
}

// Gather nearest-neighbor features; write f32 + bf16 copies
__global__ void gather_k(const float* __restrict__ src, const int* __restrict__ idx,
                         float* __restrict__ df, __bf16* __restrict__ db, int N, int C){
    long t = (long)blockIdx.x * blockDim.x + threadIdx.x;
    if (t >= (long)N * C) return;
    long r = t / C; int c = (int)(t - r * C);
    float v = src[(long)idx[r] * C + c];
    df[t] = v; db[t] = (__bf16)v;
}

// ---------------------------------------------------------------------------
// WMMA dual GEMM:  out[N,cout] = Aagg @ WrT^T + Ax @ WrootT^T + bias
// A streams: bf16 row-major, leading dim lda; weights bf16 [COpad, Kpad]
// (row = out channel). One wave (32 lanes) per 16x16 output tile.
// grid = (ceil(N/16), COpad/16), block = 32.
// ---------------------------------------------------------------------------
__global__ void wmma_dual_gemm_k(const __bf16* __restrict__ Aagg,
                                 const __bf16* __restrict__ Ax,
                                 const __bf16* __restrict__ WrT,
                                 const __bf16* __restrict__ WrootT,
                                 const float*  __restrict__ bpad,
                                 float* __restrict__ out,
                                 int N, int K, int lda, int ldw, int cout)
{
    const int lane = threadIdx.x & 31;
    const int half = lane >> 4;     // K-half select per ISA 16-bit A/B striping
    const int m16  = lane & 15;     // A row / B col / C col within tile

    long rowA = (long)blockIdx.x * 16 + m16;
    if (rowA >= N) rowA = N - 1;                 // clamp; clamped rows never stored
    const int ncol = blockIdx.y * 16 + m16;      // output channel (padded range)

    const __bf16* ar = Aagg   + rowA * (long)lda;
    const __bf16* xr = Ax     + rowA * (long)lda;
    const __bf16* wr = WrT    + (long)ncol * ldw;
    const __bf16* wt = WrootT + (long)ncol * ldw;

    v8f acc;
    {
        float bv = bpad[ncol];
        #pragma unroll
        for (int r = 0; r < 8; ++r) acc[r] = bv;   // every acc row shares column ncol
    }

    const int koff = 8 * half;
    for (int k0 = 0; k0 < K; k0 += 32){
        union U { uint4 q[2]; v16bf v; } a1, a2, b1, b2;
        const __bf16* pa = ar + k0 + koff;
        const __bf16* px = xr + k0 + koff;
        const __bf16* pw = wr + k0 + koff;
        const __bf16* pt = wt + k0 + koff;
        // lane fragment: K = k0 + 8*half + {0..7} and {16..23}  -> two b128 loads each
        a1.q[0] = *(const uint4*)(pa);  a1.q[1] = *(const uint4*)(pa + 16);
        b1.q[0] = *(const uint4*)(pw);  b1.q[1] = *(const uint4*)(pw + 16);
        a2.q[0] = *(const uint4*)(px);  a2.q[1] = *(const uint4*)(px + 16);
        b2.q[0] = *(const uint4*)(pt);  b2.q[1] = *(const uint4*)(pt + 16);

        acc = __builtin_amdgcn_wmma_f32_16x16x32_bf16(
                  false, a1.v, false, b1.v, (short)0, acc, false, false);
        acc = __builtin_amdgcn_wmma_f32_16x16x32_bf16(
                  false, a2.v, false, b2.v, (short)0, acc, false, false);
    }

    if (ncol < cout){
        long r0 = (long)blockIdx.x * 16 + 8 * half;
        #pragma unroll
        for (int r = 0; r < 8; ++r){
            long rr = r0 + r;
            if (rr < N) out[rr * (long)cout + ncol] = acc[r];
        }
    }
}

// ---------------------------------------------------------------------------
// Host orchestration
// ---------------------------------------------------------------------------
extern "C" void kernel_launch(void* const* d_in, const int* in_sizes, int n_in,
                              void* d_out, int out_size, void* d_ws, size_t ws_size,
                              hipStream_t stream)
{
    (void)n_in; (void)out_size; (void)ws_size;
    const float* x_in = (const float*)d_in[0];
    const int*   ei2  = (const int*)d_in[1];
    const int*   ei3  = (const int*)d_in[2];
    const float* pos2 = (const float*)d_in[3];
    const float* pos3 = (const float*)d_in[4];
    const float* yv   = (const float*)d_in[5];

    const int  N2 = in_sizes[0] / 4;
    const int  N3 = in_sizes[4] / 3;
    const long E2 = in_sizes[1] / 2;
    const long E3 = in_sizes[2] / 2;
    const int* src2 = ei2;           const int* dst2 = ei2 + E2;
    const int* src3 = ei3;           const int* dst3 = ei3 + E3;

    // ---- bump allocator over workspace ----
    char* ws = (char*)d_ws;
    size_t off = 0;
    auto alloc = [&](size_t bytes)->void*{
        void* p = ws + off;
        off += (bytes + 255) & ~(size_t)255;
        return p;
    };

    float*  tp2   = (float*) alloc((size_t)N2 * 3 * 4);
    float*  tp3   = (float*) alloc((size_t)N3 * 3 * 4);
    int*    nnidx = (int*)   alloc((size_t)N3 * 4);
    float*  h10   = (float*) alloc((size_t)N2 * 10 * 4);
    __bf16* hbf   = (__bf16*)alloc((size_t)N2 * 32 * 2);
    float*  ss    = (float*) alloc(512 * 4);
    const size_t bigN = (size_t)lmax((long)N2 * 256, (long)N3 * 128);
    __bf16* catbf = (__bf16*)alloc(bigN * 2);   // activation concat (bf16) / xfbf
    __bf16* aggbf = (__bf16*)alloc(bigN * 2);   // aggregated concat (bf16)
    float*  aggf  = (float*) alloc(bigN * 4);   // f32 atomic accumulation target
    float*  convo = (float*) alloc((size_t)lmax(N2, N3) * 128 * 4);
    float*  x1b   = (float*) alloc((size_t)N2 * 128 * 4);
    float*  x2b   = (float*) alloc((size_t)N2 * 128 * 4);   // contiguous with x1b
    float*  x3b   = (float*) alloc((size_t)N2 * 128 * 4);
    float*  r1b   = (float*) alloc((size_t)N2 * 128 * 4);   // contiguous with x3b
    float*  rtb   = (float*) alloc((size_t)N2 * 128 * 4);
    float*  l3b   = (float*) alloc((size_t)N2 * 128 * 4);
    float*  l1b   = (float*) alloc((size_t)N2 * 64 * 4);
    float*  l2b   = (float*) alloc((size_t)N2 * 64 * 4);

    // phase-2 aliases (N3-sized, reuse dead N2 buffers; spans are contiguous)
    float*  xf   = x1b;    // spans x1b..x2b  = N3*128 f32
    float*  xf2  = x3b;    // spans x3b..r1b  = N3*128 f32
    __bf16* xfbf = catbf;  // N3*128 bf16

    // ---- weight prep: bf16, transposed, padded ----
    struct GC {
        const float *Wr, *Wroot, *b;
        int cin, cout, Kpad, COpad;
        __bf16 *WrT, *WrootT; float* bp;
    };
    auto mkgc = [&](int iW, int cin, int cout)->GC{
        GC g;
        g.Wr    = (const float*)d_in[iW];
        g.Wroot = (const float*)d_in[iW + 1];
        g.b     = (const float*)d_in[iW + 2];
        g.cin = cin; g.cout = cout;
        g.Kpad  = ((cin  + 31) / 32) * 32;
        g.COpad = ((cout + 15) / 16) * 16;
        g.WrT    = (__bf16*)alloc((size_t)g.COpad * g.Kpad * 2);
        g.WrootT = (__bf16*)alloc((size_t)g.COpad * g.Kpad * 2);
        g.bp     = (float*) alloc((size_t)g.COpad * 4);
        int tot = g.COpad * g.Kpad;
        prep_weight_k<<<blks(tot), 256, 0, stream>>>(g.Wr,    g.b, g.WrT,    g.bp, cin, cout, g.Kpad, g.COpad);
        prep_weight_k<<<blks(tot), 256, 0, stream>>>(g.Wroot, g.b, g.WrootT, g.bp, cin, cout, g.Kpad, g.COpad);
        return g;
    };
    // params pytree leaves (sorted-key flatten), base index 6:
    // bn0.b=6 g=7 | bn2.b=8 g=9 | bn3.b=10 g=11 | conv1.Wr=12 | conv3.Wr=15
    // conv4.Wr=18 | conv5.Wr=21 | erdb1.rdb1@24 erdb1.rdb2@39 erdb2.rdb1@54 erdb2.rdb2@69
    GC gconv1 = mkgc(12, 10, 128);
    GC gconv3 = mkgc(15, 128, 128);
    GC gconv4 = mkgc(18, 128, 128);
    GC gconv5 = mkgc(21, 128, 4);
    const int rbase[4] = {24, 39, 54, 69};
    GC grdb[4][3];
    for (int i = 0; i < 4; ++i){
        grdb[i][0] = mkgc(rbase[i] + 6,  128, 64);
        grdb[i][1] = mkgc(rbase[i] + 9,  192, 64);
        grdb[i][2] = mkgc(rbase[i] + 12, 256, 128);
    }

    // ---- op lambdas ----
    auto zeroF = [&](float* p, long n){ zero_f32_k<<<blks(n), 256, 0, stream>>>(p, n); };
    auto zeroB = [&](__bf16* p, long n){ zero_bf16_k<<<blks(n), 256, 0, stream>>>(p, n); };
    auto scat  = [&](const float* f, int C, const int* s, const int* d, long E,
                     float* agg, int lda, int coff){
        scatter_add_k<<<blks(E * C), 256, 0, stream>>>(f, C, s, d, E, agg, lda, coff);
    };
    auto cvt   = [&](const float* s, int sld, int soff, __bf16* dd, int dld, int doff,
                     long rows, int cols){
        cvt_cols_k<<<blks(rows * cols), 256, 0, stream>>>(s, sld, soff, dd, dld, doff, rows, cols);
    };
    auto gemm  = [&](const GC& g, const __bf16* A, const __bf16* X, int lda,
                     float* o, int N){
        dim3 gr((N + 15) / 16, g.COpad / 16);
        wmma_dual_gemm_k<<<gr, 32, 0, stream>>>(A, X, g.WrT, g.WrootT, g.bp,
                                                o, N, g.Kpad, lda, g.Kpad, g.cout);
    };
    auto bnst = [&](const float* xx, long N, int C, int leafB){
        bn_stats_k<<<C, 256, 0, stream>>>(xx, N, C,
                                          (const float*)d_in[leafB + 1],  // g
                                          (const float*)d_in[leafB],      // b
                                          ss);
    };
    auto bnap = [&](const float* xx, float* of, __bf16* ob, int obld, int oboff,
                    long N, int C, int elu){
        bn_apply_k<<<blks(N * C), 256, 0, stream>>>(xx, ss, of, ob, obld, oboff, N, C, elu);
    };
    auto ax   = [&](const float* a, const float* bb, float al, float* o, long n){
        axpy_k<<<blks(n), 256, 0, stream>>>(a, bb, al, o, n);
    };

    // ---- RDB: agg(concat) == concat(agg) -> aggregate each block once ----
    auto run_rdb = [&](int ri, const float* xin, float* xout){
        const int base = rbase[ri];
        zeroF(aggf, (long)N2 * 256);
        cvt(xin, 128, 0, catbf, 256, 0, N2, 128);
        scat(xin, 128, src2, dst2, E2, aggf, 256, 0);
        cvt(aggf, 256, 0, aggbf, 256, 0, N2, 128);
        gemm(grdb[ri][0], aggbf, catbf, 256, convo, N2);                 // c1: K=128 -> 64
        bnst(convo, N2, 64, base + 0);                                   // bn1
        bnap(convo, l1b, catbf, 256, 128, N2, 64, 1);
        scat(l1b, 64, src2, dst2, E2, aggf, 256, 128);
        cvt(aggf, 256, 128, aggbf, 256, 128, N2, 64);
        gemm(grdb[ri][1], aggbf, catbf, 256, convo, N2);                 // c2: K=192 -> 64
        bnst(convo, N2, 64, base + 2);                                   // bn2
        bnap(convo, l2b, catbf, 256, 192, N2, 64, 1);
        scat(l2b, 64, src2, dst2, E2, aggf, 256, 192);
        cvt(aggf, 256, 192, aggbf, 256, 192, N2, 64);
        gemm(grdb[ri][2], aggbf, catbf, 256, convo, N2);                 // c3: K=256 -> 128
        bnst(convo, N2, 128, base + 4);                                  // bn3
        bnap(convo, l3b, nullptr, 0, 0, N2, 128, 1);
        ax(xin, l3b, 0.2f, xout, (long)N2 * 128);
    };
    auto run_erdb = [&](int ei, const float* xin, float* xout){
        run_rdb(2 * ei + 0, xin, r1b);
        ax(xin, r1b, 0.2f, rtb, (long)N2 * 128);
        run_rdb(2 * ei + 1, rtb, r1b);
        ax(xin, r1b, 0.2f, xout, (long)N2 * 128);
    };

    // ================= forward pass =================
    // conv1 on h = concat(x, pos2, y*ones)
    build_h_k<<<blks((long)N2 * 32), 256, 0, stream>>>(x_in, pos2, yv, h10, hbf, N2);
    zeroF(aggf, (long)N2 * 10);
    scat(h10, 10, src2, dst2, E2, aggf, 10, 0);
    zeroB(aggbf, (long)N2 * 32);
    cvt(aggf, 10, 0, aggbf, 32, 0, N2, 10);
    gemm(gconv1, aggbf, hbf, 32, convo, N2);
    bnst(convo, N2, 128, 6);                       // bn0 (no ELU)
    bnap(convo, x1b, nullptr, 0, 0, N2, 128, 0);

    run_erdb(0, x1b, x2b);                         // x2 = erdb1(x1)
    run_erdb(1, x2b, convo);                       // e2 = erdb2(x2)   (convo free here)
    ax(x1b, convo, 0.2f, x3b, (long)N2 * 128);     // x3 = x1 + 0.2*e2

    // knn1 interpolation on onera-transformed positions
    onera_k<<<blks(N2), 256, 0, stream>>>(pos2, tp2, N2);
    onera_k<<<blks(N3), 256, 0, stream>>>(pos3, tp3, N3);
    knn1_k<<<(N3 + 255) / 256, 256, 0, stream>>>(tp2, N2, tp3, N3, nnidx);
    gather_k<<<blks((long)N3 * 128), 256, 0, stream>>>(x3b, nnidx, xf, xfbf, N3, 128);

    // conv3 + bn2 + elu
    zeroF(aggf, (long)N3 * 128);
    scat(xf, 128, src3, dst3, E3, aggf, 128, 0);
    cvt(aggf, 128, 0, aggbf, 128, 0, N3, 128);
    gemm(gconv3, aggbf, xfbf, 128, convo, N3);
    bnst(convo, N3, 128, 8);
    bnap(convo, xf2, xfbf, 128, 0, N3, 128, 1);

    // conv4 + bn3 + elu
    zeroF(aggf, (long)N3 * 128);
    scat(xf2, 128, src3, dst3, E3, aggf, 128, 0);
    cvt(aggf, 128, 0, aggbf, 128, 0, N3, 128);
    gemm(gconv4, aggbf, xfbf, 128, convo, N3);
    bnst(convo, N3, 128, 10);
    bnap(convo, xf, xfbf, 128, 0, N3, 128, 1);

    // conv5 -> output [N3, 4]
    zeroF(aggf, (long)N3 * 128);
    scat(xf, 128, src3, dst3, E3, aggf, 128, 0);
    cvt(aggf, 128, 0, aggbf, 128, 0, N3, 128);
    gemm(gconv5, aggbf, xfbf, 128, (float*)d_out, N3);
}
